// Quantized_17540646437707
// MI455X (gfx1250) — matrix operand order repaired and checked by reading, not compiled, not run on hardware
//
#include <hip/hip_runtime.h>

typedef __attribute__((ext_vector_type(16))) _Float16 v16h;
typedef __attribute__((ext_vector_type(8)))  _Float16 v8h;
typedef __attribute__((ext_vector_type(4)))  _Float16 v4h;
typedef __attribute__((ext_vector_type(8)))  float    v8f;
typedef __attribute__((ext_vector_type(4)))  float    v4f;
typedef __attribute__((ext_vector_type(4)))  int      v4i;

namespace {
constexpr int kT      = 64;
constexpr int kIN     = 4096;
constexpr int kOUT    = 8192;
constexpr int kNG     = 16;
constexpr int kGS     = kIN / kNG;          // 256 elements per quant group
constexpr int kKStep  = 32;                 // WMMA f16 K
constexpr int kWaves  = 4;                  // waves per block (128 threads)
constexpr int kKSplit = 8;                  // K-dim split (atomic accumulate)
constexpr int kGPerSplit = kNG / kKSplit;   // 2 quant groups per block
constexpr int kKBlock = kGPerSplit * kGS;   // 512-wide K range per block
constexpr int kSteps  = kKBlock / kKStep;   // 16 K-steps per wave
// LDS x-tile: 64 rows x 512 halves, row padded +8 halves to spread banks
// (row stride 1040 B = 260 dwords == 4 mod 64 -> 16 rows hit disjoint banks).
constexpr int kRowStride = kKBlock + 8;     // in halves
constexpr size_t kLdsBytes = (size_t)kT * kRowStride * sizeof(_Float16); // 66,560 B
constexpr size_t kXhBytes  = (size_t)kT * kIN * sizeof(_Float16);        // 512 KB
}

__global__ void zero_out_kernel(float* __restrict__ out, int n) {
  int i = blockIdx.x * blockDim.x + threadIdx.x;
  if (i < n) out[i] = 0.0f;
}

// One-shot fp32 -> f16 conversion of x (reused by all 512 N-slices).
__global__ void cvt_x_f16_kernel(const float* __restrict__ x,
                                 _Float16* __restrict__ xh, int n4) {
  int i = blockIdx.x * blockDim.x + threadIdx.x;
  if (i < n4) {
    const v4f p = *(const v4f*)(x + (size_t)i * 4);
    v4h h;
    h[0] = (_Float16)p.x; h[1] = (_Float16)p.y;
    h[2] = (_Float16)p.z; h[3] = (_Float16)p.w;
    *(v4h*)(xh + (size_t)i * 4) = h;
  }
}

template <bool USE_XH>
__global__ __launch_bounds__(32 * kWaves)
void qgemm_wmma_f16(const float*    __restrict__ x,
                    const _Float16* __restrict__ xh,
                    const int*      __restrict__ qw,
                    const float*    __restrict__ qrange,
                    const float*    __restrict__ qmin,
                    float*          __restrict__ out) {
  extern __shared__ _Float16 xs[];          // [64][kRowStride] f16 x tile

  const int tid  = threadIdx.x;
  const int lane = tid & 31;
  const int wave = tid >> 5;

  const int ntile  = blockIdx.x * kWaves + wave;   // which 16-wide N slice
  const int obase  = ntile * 16;
  const int kbase  = blockIdx.y * kKBlock;         // K-split: this block's K range

  const int r16  = lane & 15;   // A row-in-tile / B column / C column
  const int half = lane >> 4;   // 0: lower K half, 1: upper K half

  const int  ocol = obase + r16;            // this lane's output column (B matrix)
  const long wrow = (long)ocol * kIN;       // qweight row base for that column

  // ---- Stage the block's x slice (64 x 512 f16) into LDS; shared by 4 waves.
  constexpr int kChunks    = kT * kKBlock / 8;            // 16-byte chunks
  constexpr int kFillIters = kChunks / (32 * kWaves);     // per-thread chunks
  #pragma unroll
  for (int c = 0; c < kFillIters; ++c) {
    const int idx = c * (32 * kWaves) + tid;
    const int row = idx >> 6;               // 64 chunks per row
    const int kc  = (idx & 63) * 8;
    v8h val;
    if constexpr (USE_XH) {
      val = *(const v8h*)(xh + (long)row * kIN + kbase + kc);
    } else {
      const float* xp = x + (long)row * kIN + kbase + kc;
      const v4f p0 = *(const v4f*)(xp + 0);
      const v4f p1 = *(const v4f*)(xp + 4);
      val[0] = (_Float16)p0.x; val[1] = (_Float16)p0.y;
      val[2] = (_Float16)p0.z; val[3] = (_Float16)p0.w;
      val[4] = (_Float16)p1.x; val[5] = (_Float16)p1.y;
      val[6] = (_Float16)p1.z; val[7] = (_Float16)p1.w;
    }
    *(v8h*)(xs + row * kRowStride + kc) = val;        // ds_store_b128
  }
  __syncthreads();

  // ---- Weight-stream software pipeline (HBM-bound part only). ----
  struct Stage {
    v4i   q[4];    // 16 consecutive int32 weight codes (64 B/lane)
    float scl, mn; // group dequant params for this lane's column
  };

  auto load_stage = [&](Stage& s, int t) {
    const int k0 = kbase + t * kKStep;
    const int kb = k0 + 16 * half;          // B: element e <-> K = kb + e
    s.q[0] = *(const v4i*)(qw + wrow + kb + 0);
    s.q[1] = *(const v4i*)(qw + wrow + kb + 4);
    s.q[2] = *(const v4i*)(qw + wrow + kb + 8);
    s.q[3] = *(const v4i*)(qw + wrow + kb + 12);
    __builtin_prefetch(qw + wrow + kb + 2 * kKStep, 0, 1);  // 2 steps ahead
    const int g = k0 >> 8;                  // quant group (k0 / 256)
    s.scl = qrange[ocol * kNG + g];
    s.mn  = qmin  [ocol * kNG + g];
  };

  v8f acc[4] = {};   // 64x16 tile = 4 stacked 16x16 f32 accumulators

  auto compute = [&](const Stage& s, int t) {
    // Dequantize to f16 B fragment (32x16, contiguous K: element e <-> K = 16*half + e).
    v16h b;
    #pragma unroll
    for (int e = 0; e < 4; ++e) {
      b[e +  0] = (_Float16)fmaf((float)s.q[0][e], s.scl, s.mn);
      b[e +  4] = (_Float16)fmaf((float)s.q[1][e], s.scl, s.mn);
      b[e +  8] = (_Float16)fmaf((float)s.q[2][e], s.scl, s.mn);
      b[e + 12] = (_Float16)fmaf((float)s.q[3][e], s.scl, s.mn);
    }
    // A fragments straight from LDS: lane holds K = kloc+{0..15}.
    const int kloc = t * kKStep + 16 * half;
    #pragma unroll
    for (int m = 0; m < 4; ++m) {
      const _Float16* xb = xs + (m * 16 + r16) * kRowStride + kloc;
      const v8h lo = *(const v8h*)(xb + 0);    // ds_load_b128
      const v8h hi = *(const v8h*)(xb + 8);    // ds_load_b128
      const v16h a = __builtin_shufflevector(lo, hi, 0, 1, 2, 3, 4, 5, 6, 7,
                                             8, 9, 10, 11, 12, 13, 14, 15);
      acc[m] = __builtin_amdgcn_wmma_f32_16x16x32_f16(
          /*neg_a=*/false, a, /*neg_b=*/false, b,
          /*c_mod=*/(short)0, acc[m], /*reuse_a=*/false, /*reuse_b=*/false);
    }
  };

  // Loads for step t+1 in flight while step t is dequantized and multiplied.
  Stage cur, nxt;
  load_stage(cur, 0);
  #pragma unroll 2
  for (int t = 0; t < kSteps - 1; ++t) {
    load_stage(nxt, t + 1);
    compute(cur, t);
    cur = nxt;                 // ping-pong; renamed away under unroll 2
  }
  compute(cur, kSteps - 1);

  // --- C layout: VGPR j -> M = j + 8*half, N = r16 ; accumulate K-splits ---
  #pragma unroll
  for (int m = 0; m < 4; ++m) {
    #pragma unroll
    for (int j = 0; j < 8; ++j) {
      const int t = m * 16 + j + 8 * half;
      atomicAdd(&out[(long)t * kOUT + obase + r16], acc[m][j]);
    }
  }
}

extern "C" void kernel_launch(void* const* d_in, const int* in_sizes, int n_in,
                              void* d_out, int out_size, void* d_ws, size_t ws_size,
                              hipStream_t stream) {
  const float* x      = (const float*)d_in[0];
  const int*   qw     = (const int*)d_in[1];
  const float* qrange = (const float*)d_in[2];
  const float* qmin   = (const float*)d_in[3];
  float* out = (float*)d_out;

  // Zero the output (K-split partial sums accumulate atomically).
  {
    const int n = kT * kOUT;
    zero_out_kernel<<<(n + 255) / 256, 256, 0, stream>>>(out, n);
  }

  const bool use_xh = (d_ws != nullptr) && (ws_size >= kXhBytes);
  _Float16* xh = (_Float16*)d_ws;
  if (use_xh) {
    const int n4 = (kT * kIN) / 4;   // 65536 vec4 conversions
    cvt_x_f16_kernel<<<(n4 + 255) / 256, 256, 0, stream>>>(x, xh, n4);
  }

  const int ntiles = kOUT / 16;                 // 512 N-slices
  dim3 grid(ntiles / kWaves, kKSplit);          // 128 x 8 blocks = 4096 tile-waves
  dim3 block(32 * kWaves);                      // 4 waves (wave32)
  if (use_xh) {
    qgemm_wmma_f16<true><<<grid, block, kLdsBytes, stream>>>(x, xh, qw, qrange, qmin, out);
  } else {
    qgemm_wmma_f16<false><<<grid, block, kLdsBytes, stream>>>(x, xh, qw, qrange, qmin, out);
  }
}
